// TritonBitLinear_8847632630007
// MI455X (gfx1250) — matrix operand order repaired and checked by reading, not compiled, not run on hardware
//
#include <hip/hip_runtime.h>
#include <stdint.h>

#define M_DIM 8192
#define K_DIM 4096
#define N_DIM 11008

#define BM 128
#define BN 128
#define BK 64
#define LDA 72                    // halves per LDS row (BK + 8 pad): 144B rows, 16B-aligned,
                                  // bank-stride 36 dwords -> conflict-free across 16 lanes
#define NSTAGES (K_DIM / BK)      // 64

typedef _Float16 v16h __attribute__((ext_vector_type(16)));
typedef _Float16 v8h  __attribute__((ext_vector_type(8)));
typedef float    v8f  __attribute__((ext_vector_type(8)));
typedef int      v4i  __attribute__((ext_vector_type(4)));

// async-to-LDS builtin takes pointers to int4 in global (AS1) / LDS (AS3)
typedef __attribute__((address_space(1))) v4i* gv4i_p;
typedef __attribute__((address_space(3))) v4i* lv4i_p;

#if defined(__has_builtin)
#if __has_builtin(__builtin_amdgcn_global_load_async_to_lds_b128)
#define HAVE_ASYNC_LDS 1
#endif
#endif

__device__ __forceinline__ void copy_b128_to_lds(const _Float16* g, _Float16* l) {
#ifdef HAVE_ASYNC_LDS
    // gfx1250 async global->LDS DMA (tracked by ASYNCcnt, bypasses VGPRs)
    __builtin_amdgcn_global_load_async_to_lds_b128((gv4i_p)g, (lv4i_p)l, 0, 0);
#else
    *(v8h*)l = *(const v8h*)g;   // sync fallback: global->VGPR->ds_store
#endif
}

__device__ __forceinline__ void wait_async_copies() {
#ifdef HAVE_ASYNC_LDS
#if __has_builtin(__builtin_amdgcn_s_wait_asynccnt)
    __builtin_amdgcn_s_wait_asynccnt(0);
#else
    asm volatile("s_wait_asynccnt 0" ::: "memory");
#endif
#endif
}

__global__ __launch_bounds__(256)
void w8a16_gemm_wmma(const _Float16* __restrict__ X,      // [M,K] fp16
                     const int8_t*   __restrict__ W,      // [N,K] int8
                     const float*    __restrict__ scale,  // [N]
                     const float*    __restrict__ bias,   // [N]
                     _Float16*       __restrict__ Out)    // [M,N] fp16
{
    __shared__ _Float16 sA[2][BM * LDA];   // 36 KB (fp16 activations)
    __shared__ _Float16 sB[2][BN * LDA];   // 36 KB (int8 weights converted at staging)

    const int tid   = threadIdx.x;
    const int lane  = tid & 31;
    const int wave  = tid >> 5;
    const int waveM = wave & 1;            // 0..1 -> 64-row slab
    const int waveN = wave >> 1;           // 0..3 -> 32-col slab

    const int m_blk = blockIdx.y * BM;
    const int n_blk = blockIdx.x * BN;

    v8f acc[4][2];
    const v8f vzero = {0.f, 0.f, 0.f, 0.f, 0.f, 0.f, 0.f, 0.f};
#pragma unroll
    for (int mt = 0; mt < 4; ++mt)
#pragma unroll
        for (int nt = 0; nt < 2; ++nt)
            acc[mt][nt] = vzero;

    // ---- stage: fill LDS buffer `buf` with 64-wide K-slice `s` ----
    auto stage = [&](int s, int buf) {
        const int k0 = s * BK;
        // A tile: 128x64 halves = 1024 b128 vectors, 8 per row, 4 per thread (async DMA)
#pragma unroll
        for (int v = 0; v < 4; ++v) {
            const int linear = tid + v * 256;       // 0..1023
            const int m  = linear >> 3;             // 0..127
            const int kv = linear & 7;              // 0..7 (8 halves each)
            const _Float16* g = X + (size_t)(m_blk + m) * K_DIM + k0 + kv * 8;
            _Float16*       l = &sA[buf][m * LDA + kv * 8];
            copy_b128_to_lds(g, l);
        }
        // B tile: 128x64 int8 -> fp16. Two 16-byte int8 vectors per thread.
#pragma unroll
        for (int v = 0; v < 2; ++v) {
            const int linear = tid + v * 256;       // 0..511
            const int n  = linear >> 2;             // 0..127
            const int kv = linear & 3;              // 0..3 (16 int8 each)
            const int8_t* gb = W + (size_t)(n_blk + n) * K_DIM + k0 + kv * 16;
            // prefetch the same vector two K-stages ahead into L2 (global_prefetch_b8);
            // speculative TH -> silently dropped if it runs past the allocation
            __builtin_prefetch(gb + 2 * BK, 0, 1);
            v4i raw = *(const v4i*)gb;
            const int8_t* pb = (const int8_t*)&raw;
            _Float16* lb = &sB[buf][n * LDA + kv * 16];
#pragma unroll
            for (int i = 0; i < 16; ++i)
                lb[i] = (_Float16)(short)pb[i];     // v_cvt_f16_i16, merged ds_store_b128 x2
        }
    };

    // ---- compute: 2 K-steps x 8 WMMAs on LDS buffer `buf` ----
    const int rh = lane & 15;      // row/col within 16
    const int kh = lane >> 4;      // lane-half selects K sub-block
    auto compute = [&](int buf) {
#pragma unroll
        for (int ks = 0; ks < BK; ks += 32) {
            v16h bfr[2];
#pragma unroll
            for (int nt = 0; nt < 2; ++nt) {
                const int n = waveN * 32 + nt * 16 + rh;
                // B 32x16: lanes 0-15 hold K=0..15, lanes 16-31 hold K=16..31 (contiguous)
                bfr[nt] = *(const v16h*)&sB[buf][n * LDA + ks + kh * 16];
            }
#pragma unroll
            for (int mt = 0; mt < 4; ++mt) {
                const int m = waveM * 64 + mt * 16 + rh;
                // A 16x32: lane half kh holds K = kh*16..kh*16+15 within the step (contiguous)
                v8h lo = *(const v8h*)&sA[buf][m * LDA + ks + kh * 16];
                v8h hi = *(const v8h*)&sA[buf][m * LDA + ks + kh * 16 + 8];
                v16h a;
#pragma unroll
                for (int i = 0; i < 8; ++i) { a[i] = lo[i]; a[i + 8] = hi[i]; }
#pragma unroll
                for (int nt = 0; nt < 2; ++nt)
                    acc[mt][nt] = __builtin_amdgcn_wmma_f32_16x16x32_f16(
                        false, a, false, bfr[nt],
                        (short)0, acc[mt][nt], false, false);
            }
        }
    };

    // ---- double-buffered main loop over K (64 stages, 16 WMMA/wave per barrier) ----
    stage(0, 0);
    wait_async_copies();
    __syncthreads();

    for (int s = 0; s < NSTAGES; ++s) {
        const int cur = s & 1;
        if (s + 1 < NSTAGES) stage(s + 1, cur ^ 1);
        compute(cur);
        wait_async_copies();     // next-stage A DMA landed in LDS
        __syncthreads();         // next-stage B ds_stores visible; cur free for reuse
    }

    // ---- epilogue: per-channel dequant + bias, fp16 store ----
#pragma unroll
    for (int nt = 0; nt < 2; ++nt) {
        const int n  = n_blk + waveN * 32 + nt * 16 + rh;
        const float sc = scale[n];
        const float bi = bias[n];
#pragma unroll
        for (int mt = 0; mt < 4; ++mt) {
            const int mbase = m_blk + waveM * 64 + mt * 16 + (lane >> 4) * 8;
#pragma unroll
            for (int i = 0; i < 8; ++i) {
                const float v = acc[mt][nt][i] * sc + bi;
                Out[(size_t)(mbase + i) * N_DIM + n] = (_Float16)v;
            }
        }
    }
}

extern "C" void kernel_launch(void* const* d_in, const int* in_sizes, int n_in,
                              void* d_out, int out_size, void* d_ws, size_t ws_size,
                              hipStream_t stream) {
    const _Float16* x  = (const _Float16*)d_in[0];
    const int8_t*   w  = (const int8_t*)d_in[1];
    const float*    sc = (const float*)d_in[2];
    const float*    bi = (const float*)d_in[3];
    _Float16*       out = (_Float16*)d_out;

    dim3 grid(N_DIM / BN, M_DIM / BM);   // 86 x 64
    dim3 block(256);                     // 8 wave32 waves
    w8a16_gemm_wmma<<<grid, block, 0, stream>>>(x, w, sc, bi, out);
}
